// CausalSelfAttention_24799141167437
// MI455X (gfx1250) — compile-verified, matrix operand order checked by reading, not run on hardware
//
#include <hip/hip_runtime.h>
#include <hip/hip_bf16.h>

typedef __bf16 bf16;
typedef __attribute__((ext_vector_type(16))) __bf16 v16bf;
typedef __attribute__((ext_vector_type(8)))  __bf16 v8bf;
typedef __attribute__((ext_vector_type(8)))  float  v8f;

#define B_  4
#define T_  2048
#define D_  1024
#define H_  16
#define HD_ 64
#define M_  (B_ * T_)   // 8192 token rows

// ---------------------------------------------------------------------------
// WMMA helpers (CDNA5 16x16x32 bf16 -> f32)
// A/B 16-bit fragment layout (ISA 7.12.2): lane holds row/col = lane%16; the
// 16 bf16 elements are two contiguous 8-element K-runs at K = (lane/16)*8 and
// K = 16 + (lane/16)*8.  C/D layout: lane holds N = lane%16, VGPR r holds
// M = r + 8*(lane/16).
// ---------------------------------------------------------------------------
__device__ __forceinline__ v16bf load_frag16(const bf16* base, int ld) {
    const int lane = threadIdx.x & 31;
    const int m    = lane & 15;
    const int kh   = (lane >> 4) * 8;
    const bf16* p  = base + (size_t)m * ld + kh;
    v8bf lo = *(const v8bf*)(p);
    v8bf hi = *(const v8bf*)(p + 16);
    return __builtin_shufflevector(lo, hi, 0,1,2,3,4,5,6,7,8,9,10,11,12,13,14,15);
}

__device__ __forceinline__ v8f wmma_bf16(v16bf a, v16bf b, v8f c) {
    return __builtin_amdgcn_wmma_f32_16x16x32_bf16(false, a, false, b,
                                                   (short)0, c, false, false);
}

// CDNA5 async global->LDS copy (16 bytes per lane), tracked by ASYNCcnt.
// LDS destination address = low 32 bits of the generic pointer to __shared__.
__device__ __forceinline__ void async_copy16(const bf16* gsrc, bf16* ldst) {
    unsigned lds_off = (unsigned)(size_t)ldst;
    asm volatile("global_load_async_to_lds_b128 %0, %1, off"
                 :: "v"(lds_off), "v"(gsrc) : "memory");
}
__device__ __forceinline__ void wait_async() {
    asm volatile("s_wait_asynccnt 0x0" ::: "memory");
}

// ---------------------------------------------------------------------------
// Conversion kernels
// ---------------------------------------------------------------------------
__global__ void cvt_bf16_kernel(const float* __restrict__ in,
                                bf16* __restrict__ out, int n) {
    int i = blockIdx.x * blockDim.x + threadIdx.x;
    if (i < n) out[i] = (bf16)in[i];
}

// in: [rows, cols] f32 row-major -> out: [cols, rows] bf16 row-major
__global__ void transpose_bf16_kernel(const float* __restrict__ in,
                                      bf16* __restrict__ out,
                                      int rows, int cols) {
    int i = blockIdx.x * blockDim.x + threadIdx.x;
    if (i >= rows * cols) return;
    int r = i / cols, c = i % cols;
    out[(size_t)c * rows + r] = (bf16)in[i];
}

// ---------------------------------------------------------------------------
// QKV projection: qkv = x @ w_qkv + b_qkv, scattered to Q,K (BHTD) and V^T.
// Block = 128 rows x 64 cols, 4 waves; each wave owns a 32x64 tile.
// Weight chunks double-buffered in LDS via async copy: stage(i+1) overlaps
// compute(i).  All 4 B fragments are hoisted into distinct registers so the
// 8 WMMAs of a chunk can issue back-to-back behind pipelined DS waits.
// ---------------------------------------------------------------------------
__global__ __launch_bounds__(128) void qkv_gemm_kernel(
    const bf16*  __restrict__ xb,    // [M, D]
    const bf16*  __restrict__ wT,    // [3D, D]  (w_qkv transposed)
    const float* __restrict__ bias,  // [3D]
    bf16* __restrict__ q,            // [B,H,T,HD]
    bf16* __restrict__ k,            // [B,H,T,HD]
    bf16* __restrict__ vT)           // [B,H,HD,T]
{
    const int wave = threadIdx.x >> 5;
    const int lane = threadIdx.x & 31;
    const int m0   = blockIdx.x * 128 + wave * 32;
    const int n0   = blockIdx.y * 64;

    __shared__ __align__(16) bf16 ldsW[2][64 * 32];   // double-buffered [n][k]

    auto stageW = [&](int kk, int buf) {
#pragma unroll
        for (int u = 0; u < 2; ++u) {
            const int s  = threadIdx.x * 2 + u;      // 256 x 16B segments
            const int rw = s >> 2;                   // col row 0..63
            const int ew = (s & 3) * 8;              // k elem offset
            async_copy16(wT + (size_t)(n0 + rw) * D_ + kk + ew,
                         ldsW[buf] + rw * 32 + ew);
        }
    };

    v8f acc[2][4] = {};
    const int niter = D_ / 32;
    stageW(0, 0);
    for (int i = 0; i < niter; ++i) {
        const int kk = i * 32;
        wait_async();
        __syncthreads();
        if (i + 1 < niter) stageW(kk + 32, (i + 1) & 1);

        const bf16* wbuf = ldsW[i & 1];
        v16bf a0 = load_frag16(xb + (size_t)m0 * D_ + kk, D_);
        v16bf a1 = load_frag16(xb + (size_t)(m0 + 16) * D_ + kk, D_);
        v16bf bfr[4];
#pragma unroll
        for (int t = 0; t < 4; ++t)
            bfr[t] = load_frag16(wbuf + t * 16 * 32, 32);
#pragma unroll
        for (int t = 0; t < 4; ++t) {
            acc[0][t] = wmma_bf16(a0, bfr[t], acc[0][t]);
            acc[1][t] = wmma_bf16(a1, bfr[t], acc[1][t]);
        }
    }

    const int nlo   = lane & 15;
    const int mhalf = (lane >> 4) * 8;
#pragma unroll
    for (int t = 0; t < 4; ++t) {
        const int   ncol  = n0 + t * 16 + nlo;
        const float bv    = bias[ncol];
        const int   which = ncol / D_;      // 0=q 1=k 2=v
        const int   cd    = ncol % D_;
        const int   h     = cd / HD_;
        const int   d     = cd % HD_;
#pragma unroll
        for (int mi = 0; mi < 2; ++mi) {
#pragma unroll
            for (int r = 0; r < 8; ++r) {
                const int row = m0 + mi * 16 + mhalf + r;   // global token row
                const int b   = row / T_;
                const int tt  = row % T_;
                const bf16 val = (bf16)(acc[mi][t][r] + bv);
                if (which == 0)
                    q[(((size_t)b * H_ + h) * T_ + tt) * HD_ + d] = val;
                else if (which == 1)
                    k[(((size_t)b * H_ + h) * T_ + tt) * HD_ + d] = val;
                else
                    vT[(((size_t)b * H_ + h) * HD_ + d) * T_ + tt] = val;
            }
        }
    }
}

// ---------------------------------------------------------------------------
// Causal flash attention.  Block = 4 waves = 64 q rows of one (b,h); all
// waves stream the SAME 32-key chunks, double-buffered in LDS via async
// copies so stage(i+1) overlaps compute(i).  K and V fragments are hoisted
// into distinct registers ahead of their WMMA bursts.
// ---------------------------------------------------------------------------
__global__ __launch_bounds__(128) void attn_kernel(
    const bf16* __restrict__ q,   // [B,H,T,HD]
    const bf16* __restrict__ km,  // [B,H,T,HD]
    const bf16* __restrict__ vT,  // [B,H,HD,T]
    bf16* __restrict__ o)         // [M, D], col = h*HD + d
{
    const int wave  = threadIdx.x >> 5;
    const int lane  = threadIdx.x & 31;
    const int bh    = blockIdx.y;          // b*H + h
    const int bb    = bh / H_;
    const int hh    = bh % H_;
    const int qbase = blockIdx.x * 64;
    const int q0    = qbase + wave * 16;

    const bf16* qp = q  + (size_t)bh * T_ * HD_;
    const bf16* kp = km + (size_t)bh * T_ * HD_;
    const bf16* vp = vT + (size_t)bh * HD_ * T_;

    __shared__ __align__(16) bf16 ldsK[2][32 * HD_];   // [key][d]  2 x 4 KB
    __shared__ __align__(16) bf16 ldsV[2][HD_ * 32];   // [d][key]  2 x 4 KB
    __shared__ __align__(16) bf16 lds_p[4][16 * 32];   // per-wave P tiles
    bf16* pbuf = lds_p[wave];

    auto stageKV = [&](int kb, int buf) {
#pragma unroll
        for (int u = 0; u < 2; ++u) {
            const int s  = threadIdx.x * 2 + u;          // 256 segments each
            const int rk = s >> 3;                       // key row 0..31
            const int ek = (s & 7) * 8;                  // d elem offset
            async_copy16(kp + (size_t)(kb + rk) * HD_ + ek,
                         ldsK[buf] + rk * HD_ + ek);
            const int rv = s >> 2;                       // d row 0..63
            const int ev = (s & 3) * 8;                  // key elem offset
            async_copy16(vp + (size_t)rv * T_ + kb + ev,
                         ldsV[buf] + rv * 32 + ev);
        }
    };

    // Q fragments, resident for the whole KV sweep (d-chunks 0..31, 32..63)
    const v16bf qa0 = load_frag16(qp + (size_t)q0 * HD_,      HD_);
    const v16bf qa1 = load_frag16(qp + (size_t)q0 * HD_ + 32, HD_);

    v8f   oacc[4] = {};
    float mrow[8], lrow[8], scrow[8];
#pragma unroll
    for (int r = 0; r < 8; ++r) { mrow[r] = -__builtin_inff(); lrow[r] = 0.f; }

    const float scale = 0.03125f;                 // 1/sqrt(D) = 1/32
    const int   nlo   = lane & 15;
    const int   mhalf = (lane >> 4) * 8;

    const int nchunks = qbase / 32 + 2;           // covers keys 0..qbase+63
    stageKV(0, 0);
    for (int i = 0; i < nchunks; ++i) {
        const int kb = i * 32;
        wait_async();
        __syncthreads();
        if (i + 1 < nchunks) stageKV(kb + 32, (i + 1) & 1);

        const bf16* Kb = ldsK[i & 1];
        const bf16* Vb = ldsV[i & 1];

        // ---- scores: two 16x16 tiles over 32 keys (K frags from LDS) ------
        v16bf kf[4];
        kf[0] = load_frag16(Kb,                 HD_);
        kf[1] = load_frag16(Kb + 32,            HD_);
        kf[2] = load_frag16(Kb + 16 * HD_,      HD_);
        kf[3] = load_frag16(Kb + 16 * HD_ + 32, HD_);
        v8f s0 = {}, s1 = {};
        s0 = wmma_bf16(qa0, kf[0], s0);
        s0 = wmma_bf16(qa1, kf[1], s0);
        s1 = wmma_bf16(qa0, kf[2], s1);
        s1 = wmma_bf16(qa1, kf[3], s1);

        // ---- causal mask + online softmax (per-row, half-wave reduce) -----
#pragma unroll
        for (int r = 0; r < 8; ++r) {
            const int qg = q0 + mhalf + r;
            float v0 = s0[r] * scale;
            float v1 = s1[r] * scale;
            if (kb + nlo      > qg) v0 = -__builtin_inff();
            if (kb + 16 + nlo > qg) v1 = -__builtin_inff();

            float mx = fmaxf(v0, v1);
            mx = fmaxf(mx, __shfl_xor(mx, 1, 32));
            mx = fmaxf(mx, __shfl_xor(mx, 2, 32));
            mx = fmaxf(mx, __shfl_xor(mx, 4, 32));
            mx = fmaxf(mx, __shfl_xor(mx, 8, 32));

            // chunk 0 is never fully masked, so mrow is finite afterwards and
            // fully-masked later chunks degenerate to p=0, sc=1 (no NaNs).
            const float mnew = fmaxf(mrow[r], mx);
            const float p0 = __expf(v0 - mnew);
            const float p1 = __expf(v1 - mnew);
            const float sc = __expf(mrow[r] - mnew);

            float rs = p0 + p1;
            rs += __shfl_xor(rs, 1, 32);
            rs += __shfl_xor(rs, 2, 32);
            rs += __shfl_xor(rs, 4, 32);
            rs += __shfl_xor(rs, 8, 32);

            lrow[r]  = lrow[r] * sc + rs;
            mrow[r]  = mnew;
            scrow[r] = sc;

            // stage P tile (C-layout -> row-major 16x32 in LDS)
            pbuf[(mhalf + r) * 32 + nlo]      = (bf16)p0;
            pbuf[(mhalf + r) * 32 + 16 + nlo] = (bf16)p1;
        }

        // rescale running output accumulators
#pragma unroll
        for (int t = 0; t < 4; ++t)
#pragma unroll
            for (int r = 0; r < 8; ++r) oacc[t][r] *= scrow[r];

        // intra-wave LDS transpose handoff: DS ops are in-order per wave
        asm volatile("s_wait_dscnt 0x0" ::: "memory");
        const v16bf pa = load_frag16(pbuf, 32);

        // ---- PV: P[16x32] x V[32x64] as four 16x16 tiles (V from LDS) -----
        v16bf vf[4];
#pragma unroll
        for (int t = 0; t < 4; ++t)
            vf[t] = load_frag16(Vb + t * 16 * 32, 32);
#pragma unroll
        for (int t = 0; t < 4; ++t)
            oacc[t] = wmma_bf16(pa, vf[t], oacc[t]);
    }

    // ---- normalize and emit bf16 attention output -------------------------
#pragma unroll
    for (int t = 0; t < 4; ++t) {
#pragma unroll
        for (int r = 0; r < 8; ++r) {
            const int row = q0 + mhalf + r;
            const int col = hh * HD_ + t * 16 + nlo;
            const float val = oacc[t][r] / lrow[r];
            o[((size_t)bb * T_ + row) * D_ + col] = (bf16)val;
        }
    }
}

// ---------------------------------------------------------------------------
// Output projection: out = o @ w_out + b_out (fp32 result).
// Same blocking as QKV GEMM: 128x64 block, 32x64 per wave, double-buffered
// weight staging via async-to-LDS, hoisted B fragments.
// ---------------------------------------------------------------------------
__global__ __launch_bounds__(128) void out_gemm_kernel(
    const bf16*  __restrict__ ob,    // [M, D]
    const bf16*  __restrict__ wT,    // [D, D]  (w_out transposed)
    const float* __restrict__ bias,  // [D]
    float* __restrict__ out)         // [M, D]
{
    const int wave = threadIdx.x >> 5;
    const int lane = threadIdx.x & 31;
    const int m0   = blockIdx.x * 128 + wave * 32;
    const int n0   = blockIdx.y * 64;

    __shared__ __align__(16) bf16 ldsW[2][64 * 32];

    auto stageW = [&](int kk, int buf) {
#pragma unroll
        for (int u = 0; u < 2; ++u) {
            const int s  = threadIdx.x * 2 + u;
            const int rw = s >> 2;
            const int ew = (s & 3) * 8;
            async_copy16(wT + (size_t)(n0 + rw) * D_ + kk + ew,
                         ldsW[buf] + rw * 32 + ew);
        }
    };

    v8f acc[2][4] = {};
    const int niter = D_ / 32;
    stageW(0, 0);
    for (int i = 0; i < niter; ++i) {
        const int kk = i * 32;
        wait_async();
        __syncthreads();
        if (i + 1 < niter) stageW(kk + 32, (i + 1) & 1);

        const bf16* wbuf = ldsW[i & 1];
        v16bf a0 = load_frag16(ob + (size_t)m0 * D_ + kk, D_);
        v16bf a1 = load_frag16(ob + (size_t)(m0 + 16) * D_ + kk, D_);
        v16bf bfr[4];
#pragma unroll
        for (int t = 0; t < 4; ++t)
            bfr[t] = load_frag16(wbuf + t * 16 * 32, 32);
#pragma unroll
        for (int t = 0; t < 4; ++t) {
            acc[0][t] = wmma_bf16(a0, bfr[t], acc[0][t]);
            acc[1][t] = wmma_bf16(a1, bfr[t], acc[1][t]);
        }
    }

    const int nlo   = lane & 15;
    const int mhalf = (lane >> 4) * 8;
#pragma unroll
    for (int t = 0; t < 4; ++t) {
        const int   ncol = n0 + t * 16 + nlo;
        const float bv   = bias[ncol];
#pragma unroll
        for (int mi = 0; mi < 2; ++mi) {
#pragma unroll
            for (int r = 0; r < 8; ++r) {
                const int row = m0 + mi * 16 + mhalf + r;
                out[(size_t)row * D_ + ncol] = acc[mi][t][r] + bv;
            }
        }
    }
}

// ---------------------------------------------------------------------------
// Launch
// ---------------------------------------------------------------------------
extern "C" void kernel_launch(void* const* d_in, const int* in_sizes, int n_in,
                              void* d_out, int out_size, void* d_ws, size_t ws_size,
                              hipStream_t stream) {
    const float* x     = (const float*)d_in[0];  // [B,T,D]
    const float* w_qkv = (const float*)d_in[1];  // [D, 3D]
    const float* b_qkv = (const float*)d_in[2];  // [3D]
    const float* w_out = (const float*)d_in[3];  // [D, D]
    const float* b_out = (const float*)d_in[4];  // [D]
    float*       out   = (float*)d_out;          // [B,T,D]

    // workspace carve-up (bf16 buffers), 16B-aligned sections
    char* ws = (char*)d_ws;
    bf16* xb    = (bf16*)ws;                 ws += (size_t)M_ * D_ * 2;       // 16 MB
    bf16* wqkvT = (bf16*)ws;                 ws += (size_t)3 * D_ * D_ * 2;   //  6 MB
    bf16* woutT = (bf16*)ws;                 ws += (size_t)D_ * D_ * 2;       //  2 MB
    bf16* qb    = (bf16*)ws;                 ws += (size_t)M_ * D_ * 2;       // 16 MB
    bf16* kb    = (bf16*)ws;                 ws += (size_t)M_ * D_ * 2;       // 16 MB
    bf16* vTb   = (bf16*)ws;                 ws += (size_t)M_ * D_ * 2;       // 16 MB
    bf16* obuf  = (bf16*)ws;                 ws += (size_t)M_ * D_ * 2;       // 16 MB

    // 1) precision conversion + weight transposes
    {
        int n = M_ * D_;
        cvt_bf16_kernel<<<(n + 255) / 256, 256, 0, stream>>>(x, xb, n);
    }
    {
        int n = D_ * 3 * D_;
        transpose_bf16_kernel<<<(n + 255) / 256, 256, 0, stream>>>(w_qkv, wqkvT, D_, 3 * D_);
    }
    {
        int n = D_ * D_;
        transpose_bf16_kernel<<<(n + 255) / 256, 256, 0, stream>>>(w_out, woutT, D_, D_);
    }

    // 2) QKV projection GEMM (weights double-buffered via async-to-LDS)
    qkv_gemm_kernel<<<dim3(M_ / 128, (3 * D_) / 64), 128, 0, stream>>>(
        xb, wqkvT, b_qkv, qb, kb, vTb);

    // 3) causal flash attention (K/V double-buffered via async-to-LDS)
    attn_kernel<<<dim3(T_ / 64, B_ * H_), 128, 0, stream>>>(qb, kb, vTb, obuf);

    // 4) output projection GEMM (fp32 out + bias)
    out_gemm_kernel<<<dim3(M_ / 128, D_ / 64), 128, 0, stream>>>(
        obuf, woutT, b_out, out);
}
